// PairformerStack_53377853555085
// MI455X (gfx1250) — compile-verified, matrix operand order checked by reading, not run on hardware
//
#include <hip/hip_runtime.h>
#include <hip/hip_bf16.h>
#include <stdint.h>

// ---------------------------------------------------------------------------
// Pairformer stack (AF3) for MI455X / gfx1250.
// All matmuls routed through v_wmma_f32_16x16x32_bf16 (wave32 WMMA).
// ---------------------------------------------------------------------------

typedef __attribute__((ext_vector_type(16))) __bf16 v16bf;
typedef __attribute__((ext_vector_type(8)))  float  v8f;

#define Bc   2
#define Nc   160
#define CPc  128
#define CTc  128
#define CSc  384
#define HTc  4
#define DTc  32
#define HSc  16
#define DSc  32
static const long MPr = (long)Bc * Nc * Nc;   // 51200 pair rows
static const long MSr = (long)Bc * Nc;        // 320 single rows

union FragB { v16bf v; uint4 q[2]; __bf16 h[16]; };

__device__ inline v8f v8zero() {
  v8f z = {0.f,0.f,0.f,0.f,0.f,0.f,0.f,0.f};
  return z;
}

__device__ inline v8f wmma_bf16(v16bf a, v16bf b, v8f c) {
  // D = A(16x32 bf16) * B(32x16 bf16) + C(16x16 f32)
  return __builtin_amdgcn_wmma_f32_16x16x32_bf16(
      false, a, false, b, (short)0, c, false, false);
}

// K(e) per ISA 16-bit A-matrix 16x32 layout: lanes 0-15 row M=lane,
// VGPR 0-3 => K 0..7 (+8 for hi half-wave), VGPR 4-7 => K 16..23 (+8).
__device__ inline v16bf load_frag_rows(const __bf16* base, int stride, int lane) {
  int r = lane & 15, hi = lane >> 4;
  FragB f;
  const __bf16* p = base + (long)r * stride;
  f.q[0] = *reinterpret_cast<const uint4*>(p + 8 * hi);
  f.q[1] = *reinterpret_cast<const uint4*>(p + 16 + 8 * hi);
  return f.v;
}

__device__ inline v16bf load_frag_rows_f32(const float* base, int stride, int lane) {
  int r = lane & 15, hi = lane >> 4;
  FragB f;
  const float* p = base + (long)r * stride;
#pragma unroll
  for (int e = 0; e < 16; ++e) {
    int kk = (e < 8 ? e : e + 8) + 8 * hi;
    f.h[e] = (__bf16)p[kk];
  }
  return f.v;
}

__device__ inline v16bf load_frag_bcol(const __bf16* base, int stride, int lane) {
  int c = lane & 15, hi = lane >> 4;
  FragB f;
#pragma unroll
  for (int e = 0; e < 16; ++e) {
    int kk = (e < 8 ? e : e + 8) + 8 * hi;
    f.h[e] = base[(long)kk * stride + c];
  }
  return f.v;
}

__device__ inline float sigmoidf_(float x) { return 1.f / (1.f + __expf(-x)); }

// ---------------------------------------------------------------------------
// LayerNorm (f32 in -> bf16 out), one wave per row; optional (i,j) transpose
// of the source row index for the ending-node triangle attention.
// ---------------------------------------------------------------------------
__global__ __launch_bounds__(128)
void k_layernorm(const float* __restrict__ x, const float* __restrict__ g,
                 const float* __restrict__ b, __bf16* __restrict__ out,
                 int rows, int C, int Nsz, int transposeNN) {
  int row = blockIdx.x * 4 + (threadIdx.x >> 5);
  int lane = threadIdx.x & 31;
  if (row >= rows) return;
  long src = row;
  if (transposeNN) {
    int bb = row / (Nsz * Nsz);
    int rem = row % (Nsz * Nsz);
    int i = rem / Nsz, j = rem % Nsz;
    src = ((long)bb * Nsz + j) * Nsz + i;
  }
  const float* px = x + src * (long)C;
  float s = 0.f, ss = 0.f;
  for (int c = lane; c < C; c += 32) { float v = px[c]; s += v; ss += v * v; }
#pragma unroll
  for (int o = 16; o > 0; o >>= 1) {
    s  += __shfl_xor(s,  o, 32);
    ss += __shfl_xor(ss, o, 32);
  }
  float mean = s / C;
  float var  = ss / C - mean * mean;
  float inv  = rsqrtf(var + 1e-5f);
  __bf16* po = out + (long)row * C;
  for (int c = lane; c < C; c += 32)
    po[c] = (__bf16)((px[c] - mean) * inv * g[c] + b[c]);
}

// ---------------------------------------------------------------------------
// Generic WMMA GEMM: C[M,N] = A[M,K](bf16) * W[K,N], 64x64 tile, 4 waves.
// EPI: 0 none, 1 sigmoid, 2 sigmoid(AW1)*(AW2), 3 silu(AW1)*(AW2)
// W16: weights provided as bf16 (batched triangle einsum) instead of f32.
// Optional elementwise gate-multiply (mulsrc) and residual add (res).
// ---------------------------------------------------------------------------
template <int EPI, bool W16>
__global__ __launch_bounds__(128)
void k_gemm(const __bf16* __restrict__ A,
            const float*  __restrict__ Wf, const __bf16* __restrict__ Wh,
            const float*  __restrict__ W2f,
            float* __restrict__ outF, __bf16* __restrict__ outH,
            const float* __restrict__ mulsrc, const float* __restrict__ res,
            int M, int N, int K, long sA, long sW, long sC) {
  __shared__ __bf16 As[64][40];
  __shared__ __bf16 Bs[64][40];   // transposed: Bs[n][k]
  __shared__ __bf16 Bs2[64][40];
  const int tid = threadIdx.x;
  const int wv = tid >> 5, lane = tid & 31;
  const int mb = blockIdx.y * 64, nb = blockIdx.x * 64;
  const long zA = (long)blockIdx.z * sA;
  const long zW = (long)blockIdx.z * sW;
  const long zC = (long)blockIdx.z * sC;
  const int wr = (wv >> 1) * 32, wc = (wv & 1) * 32;

  v8f acc[2][2], acc2[2][2];
#pragma unroll
  for (int i = 0; i < 2; ++i)
#pragma unroll
    for (int j = 0; j < 2; ++j) { acc[i][j] = v8zero(); acc2[i][j] = v8zero(); }

  for (int k0 = 0; k0 < K; k0 += 32) {
    // stage A tile (64 rows x 32 bf16), 16B vector loads
    for (int t = tid; t < 256; t += 128) {
      int r = t >> 2, ch = t & 3;
      int gr = mb + r;
      uint4 val = {0u, 0u, 0u, 0u};
      if (gr < M)
        val = *reinterpret_cast<const uint4*>(A + zA + (long)gr * K + k0 + ch * 8);
      *reinterpret_cast<uint4*>(&As[r][ch * 8]) = val;
    }
    // stage W tile transposed (convert f32->bf16 on the fly unless W16)
    for (int t = tid; t < 2048; t += 128) {
      int kk = t >> 6, c = t & 63;
      int gc = nb + c;
      float v1 = 0.f, v2 = 0.f;
      if (gc < N) {
        long wi = zW + (long)(k0 + kk) * N + gc;
        if constexpr (W16) v1 = (float)Wh[wi];
        else               v1 = Wf[wi];
        if constexpr (EPI >= 2) v2 = W2f[wi];
      }
      Bs[c][kk] = (__bf16)v1;
      if constexpr (EPI >= 2) Bs2[c][kk] = (__bf16)v2;
    }
    if (lane == 0 && k0 + 32 < K)   // hint next A tile toward the caches
      __builtin_prefetch(A + zA + (long)(mb + wr) * K + k0 + 32, 0, 1);
    __syncthreads();

    v16bf af[2], bfr[2], bfr2[2];
    af[0]  = load_frag_rows(&As[wr][0],      40, lane);
    af[1]  = load_frag_rows(&As[wr + 16][0], 40, lane);
    bfr[0] = load_frag_rows(&Bs[wc][0],      40, lane);
    bfr[1] = load_frag_rows(&Bs[wc + 16][0], 40, lane);
    if constexpr (EPI >= 2) {
      bfr2[0] = load_frag_rows(&Bs2[wc][0],      40, lane);
      bfr2[1] = load_frag_rows(&Bs2[wc + 16][0], 40, lane);
    }
#pragma unroll
    for (int mi = 0; mi < 2; ++mi)
#pragma unroll
      for (int ni = 0; ni < 2; ++ni) {
        acc[mi][ni] = wmma_bf16(af[mi], bfr[ni], acc[mi][ni]);
        if constexpr (EPI >= 2)
          acc2[mi][ni] = wmma_bf16(af[mi], bfr2[ni], acc2[mi][ni]);
      }
    __syncthreads();
  }

  // epilogue (C layout: VGPR v -> row v + 8*(lane>>4), col lane&15)
#pragma unroll
  for (int mi = 0; mi < 2; ++mi)
#pragma unroll
    for (int ni = 0; ni < 2; ++ni)
#pragma unroll
      for (int vv = 0; vv < 8; ++vv) {
        int row = mb + wr + mi * 16 + vv + 8 * (lane >> 4);
        int col = nb + wc + ni * 16 + (lane & 15);
        if (row < M && col < N) {
          float x = acc[mi][ni][vv];
          float y;
          if constexpr (EPI == 0)      y = x;
          else if constexpr (EPI == 1) y = sigmoidf_(x);
          else if constexpr (EPI == 2) y = sigmoidf_(x) * acc2[mi][ni][vv];
          else                         y = (x * sigmoidf_(x)) * acc2[mi][ni][vv];
          long idx = zC + (long)row * N + col;
          if (mulsrc) y *= mulsrc[idx];
          if (res)    y += res[idx];
          if (outF) outF[idx] = y;
          if (outH) outH[idx] = (__bf16)y;
        }
      }
}

// ---------------------------------------------------------------------------
// Triangle-einsum repack: [b,i,j,c] bf16 -> [(b*CT+c), p, q] bf16.
// swap selects (p,q)->(q,p) source mapping (covers all 4 in/out cases).
// ---------------------------------------------------------------------------
__global__ void k_pack_tri(const __bf16* __restrict__ src,
                           __bf16* __restrict__ dst, int swap) {
  long t = (long)blockIdx.x * 256 + threadIdx.x;
  if (t >= 6553600L) return;
  int qq = (int)(t % 160); long r = t / 160;
  int pp = (int)(r % 160); r /= 160;
  int c  = (int)(r % 128); int b = (int)(r / 128);
  int rr = swap ? qq : pp;
  int ss = swap ? pp : qq;
  dst[t] = src[(((long)(b * 160 + rr)) * 160 + ss) * 128 + c];
}

__global__ void k_unpack_tri(const float* __restrict__ src,
                             float* __restrict__ dst) {
  long t = (long)blockIdx.x * 256 + threadIdx.x;
  if (t >= 6553600L) return;
  int c = (int)(t % 128); long r = t / 128;
  int j = (int)(r % 160); r /= 160;
  int i = (int)(r % 160); int b = (int)(r / 160);
  dst[t] = src[(((long)(b * 128 + c)) * 160 + i) * 160 + j];
}

__global__ void k_add_transpose(float* __restrict__ z,
                                const float* __restrict__ x) {
  long t = (long)blockIdx.x * 256 + threadIdx.x;
  if (t >= 6553600L) return;
  int c = (int)(t % 128); long r = t / 128;
  int j = (int)(r % 160); r /= 160;
  int i = (int)(r % 160); int b = (int)(r / 160);
  z[t] += x[(((long)(b * 160 + j)) * 160 + i) * 128 + c];
}

__global__ void k_mul_to_bf16(const float* __restrict__ a,
                              const float* __restrict__ g,
                              __bf16* __restrict__ o, long n) {
  long t = (long)blockIdx.x * 256 + threadIdx.x;
  if (t >= n) return;
  o[t] = (__bf16)(a[t] * g[t]);
}

// ---------------------------------------------------------------------------
// Triangle attention (starting-node form). One WG = (b,h,i, query-tile jt).
// Q[16,32] x K[160,32]^T via one WMMA per 16-col tile (D=32 = one K-chunk),
// softmax over 160 keys in LDS, then P[16,160] x V[160,32] via 5 WMMA chunks.
// ---------------------------------------------------------------------------
__global__ __launch_bounds__(64)
void k_tri_attn(const __bf16* __restrict__ q, const __bf16* __restrict__ k,
                const __bf16* __restrict__ v, const float* __restrict__ bias,
                float* __restrict__ o) {
  int jt = blockIdx.x;            // 0..9
  int i  = blockIdx.y;            // 0..159
  int bh = blockIdx.z;            // b*4+h
  int b = bh >> 2, h = bh & 3;
  __shared__ __bf16 Qs[16][32];
  __shared__ __bf16 Ks[160][32];
  __shared__ __bf16 Vs[160][32];
  __shared__ float  Ss[16][164];
  __shared__ float  red[16][4];
  int tid = threadIdx.x, wv = tid >> 5, lane = tid & 31;

  for (int t = tid; t < 16 * 8; t += 64) {
    int r = t >> 3, ch = t & 7;
    long off = (((long)(b * 160 + i)) * 160 + (jt * 16 + r)) * 128 + h * 32 + ch * 4;
    *reinterpret_cast<uint2*>(&Qs[r][ch * 4]) =
        *reinterpret_cast<const uint2*>(q + off);
  }
  for (int t = tid; t < 160 * 8; t += 64) {
    int r = t >> 3, ch = t & 7;
    long off = (((long)(b * 160 + i)) * 160 + r) * 128 + h * 32 + ch * 4;
    *reinterpret_cast<uint2*>(&Ks[r][ch * 4]) =
        *reinterpret_cast<const uint2*>(k + off);
    *reinterpret_cast<uint2*>(&Vs[r][ch * 4]) =
        *reinterpret_cast<const uint2*>(v + off);
  }
  __syncthreads();

  const float scale = 0.17677669529663687f;  // 1/sqrt(32)
  for (int ct = wv; ct < 10; ct += 2) {
    v16bf aq = load_frag_rows(&Qs[0][0], 32, lane);
    v16bf bk = load_frag_rows(&Ks[ct * 16][0], 32, lane);
    v8f acc = wmma_bf16(aq, bk, v8zero());
    int col = ct * 16 + (lane & 15);
#pragma unroll
    for (int vv = 0; vv < 8; ++vv) {
      int m = vv + 8 * (lane >> 4);
      int jrow = jt * 16 + m;
      float val = acc[vv] * scale +
                  bias[(((long)(b * 160 + jrow)) * 160 + col) * 4 + h];
      Ss[m][col] = val;
    }
  }
  __syncthreads();

  // softmax over 160 keys: 4 threads per row
  int row = tid >> 2, part = tid & 3;
  float mx = -3.0e38f;
  for (int c = part; c < 160; c += 4) mx = fmaxf(mx, Ss[row][c]);
  red[row][part] = mx;
  __syncthreads();
  mx = fmaxf(fmaxf(red[row][0], red[row][1]), fmaxf(red[row][2], red[row][3]));
  float sm = 0.f;
  for (int c = part; c < 160; c += 4) {
    float e = __expf(Ss[row][c] - mx);
    Ss[row][c] = e;
    sm += e;
  }
  __syncthreads();
  red[row][part] = sm;
  __syncthreads();
  sm = red[row][0] + red[row][1] + red[row][2] + red[row][3];
  float invs = 1.f / sm;
  for (int c = part; c < 160; c += 4) Ss[row][c] *= invs;
  __syncthreads();

  // O = P * V : wave wv owns d-columns [wv*16, wv*16+16)
  v8f accO = v8zero();
  for (int kc = 0; kc < 5; ++kc) {
    v16bf ap = load_frag_rows_f32(&Ss[0][kc * 32], 164, lane);
    v16bf bv = load_frag_bcol(&Vs[kc * 32][0] + wv * 16, 32, lane);
    accO = wmma_bf16(ap, bv, accO);
  }
#pragma unroll
  for (int vv = 0; vv < 8; ++vv) {
    int m = vv + 8 * (lane >> 4);
    int jrow = jt * 16 + m;
    int d = wv * 16 + (lane & 15);
    o[(((long)(b * 160 + i)) * 160 + jrow) * 128 + h * 32 + d] = accO[vv];
  }
}

// ---------------------------------------------------------------------------
// Single-rep attention with pair bias + mask. One WG = (b,h, query-tile it).
// ---------------------------------------------------------------------------
__global__ __launch_bounds__(64)
void k_single_attn(const __bf16* __restrict__ q, const __bf16* __restrict__ k,
                   const __bf16* __restrict__ v, const float* __restrict__ bias,
                   const unsigned char* __restrict__ mask,
                   float* __restrict__ o) {
  int it = blockIdx.x;            // 0..9
  int bh = blockIdx.z;            // b*16+h
  int b = bh >> 4, h = bh & 15;
  __shared__ __bf16 Qs[16][32];
  __shared__ __bf16 Ks[160][32];
  __shared__ __bf16 Vs[160][32];
  __shared__ float  Ss[16][164];
  __shared__ float  red[16][4];
  int tid = threadIdx.x, wv = tid >> 5, lane = tid & 31;

  for (int t = tid; t < 16 * 8; t += 64) {
    int r = t >> 3, ch = t & 7;
    long off = ((long)(b * 160 + it * 16 + r)) * 512 + h * 32 + ch * 4;
    *reinterpret_cast<uint2*>(&Qs[r][ch * 4]) =
        *reinterpret_cast<const uint2*>(q + off);
  }
  for (int t = tid; t < 160 * 8; t += 64) {
    int r = t >> 3, ch = t & 7;
    long off = ((long)(b * 160 + r)) * 512 + h * 32 + ch * 4;
    *reinterpret_cast<uint2*>(&Ks[r][ch * 4]) =
        *reinterpret_cast<const uint2*>(k + off);
    *reinterpret_cast<uint2*>(&Vs[r][ch * 4]) =
        *reinterpret_cast<const uint2*>(v + off);
  }
  __syncthreads();

  const float scale = 0.17677669529663687f;
  for (int ct = wv; ct < 10; ct += 2) {
    v16bf aq = load_frag_rows(&Qs[0][0], 32, lane);
    v16bf bk = load_frag_rows(&Ks[ct * 16][0], 32, lane);
    v8f acc = wmma_bf16(aq, bk, v8zero());
    int col = ct * 16 + (lane & 15);
#pragma unroll
    for (int vv = 0; vv < 8; ++vv) {
      int m = vv + 8 * (lane >> 4);
      int irow = it * 16 + m;
      long pij = ((long)(b * 160 + irow)) * 160 + col;
      float val = acc[vv] * scale + bias[pij * 16 + h] +
                  (mask[pij] ? 0.f : -1e9f);
      Ss[m][col] = val;
    }
  }
  __syncthreads();

  int row = tid >> 2, part = tid & 3;
  float mx = -3.0e38f;
  for (int c = part; c < 160; c += 4) mx = fmaxf(mx, Ss[row][c]);
  red[row][part] = mx;
  __syncthreads();
  mx = fmaxf(fmaxf(red[row][0], red[row][1]), fmaxf(red[row][2], red[row][3]));
  float sm = 0.f;
  for (int c = part; c < 160; c += 4) {
    float e = __expf(Ss[row][c] - mx);
    Ss[row][c] = e;
    sm += e;
  }
  __syncthreads();
  red[row][part] = sm;
  __syncthreads();
  sm = red[row][0] + red[row][1] + red[row][2] + red[row][3];
  float invs = 1.f / sm;
  for (int c = part; c < 160; c += 4) Ss[row][c] *= invs;
  __syncthreads();

  v8f accO = v8zero();
  for (int kc = 0; kc < 5; ++kc) {
    v16bf ap = load_frag_rows_f32(&Ss[0][kc * 32], 164, lane);
    v16bf bv = load_frag_bcol(&Vs[kc * 32][0] + wv * 16, 32, lane);
    accO = wmma_bf16(ap, bv, accO);
  }
#pragma unroll
  for (int vv = 0; vv < 8; ++vv) {
    int m = vv + 8 * (lane >> 4);
    int irow = it * 16 + m;
    int d = wv * 16 + (lane & 15);
    o[((long)(b * 160 + irow)) * 512 + h * 32 + d] = accO[vv];
  }
}

// ---------------------------------------------------------------------------
// Host-side orchestration
// ---------------------------------------------------------------------------

// Param order = sorted(param dict keys); per-block element counts.
enum {
  I_pt_W1 = 0, I_pt_W2, I_pt_W3, I_pt_lnb, I_pt_lng,
  I_sa_Wb, I_sa_Wg, I_sa_Wk, I_sa_Wo, I_sa_Wq, I_sa_Wv,
  I_sa_lnsb, I_sa_lnsg, I_sa_lnzb, I_sa_lnzg,
  I_st_W1, I_st_W2, I_st_W3, I_st_lnb, I_st_lng,
  I_tae_Wb, I_tae_Wg, I_tae_Wk, I_tae_Wo, I_tae_Wq, I_tae_Wv, I_tae_lnb, I_tae_lng,
  I_tas_Wb, I_tas_Wg, I_tas_Wk, I_tas_Wo, I_tas_Wq, I_tas_Wv, I_tas_lnb, I_tas_lng,
  I_tmi_Wag, I_tmi_Wap, I_tmi_Wbg, I_tmi_Wbp, I_tmi_Wg, I_tmi_Wo,
  I_tmi_ln2b, I_tmi_ln2g, I_tmi_lnb, I_tmi_lng,
  I_tmo_Wag, I_tmo_Wap, I_tmo_Wbg, I_tmo_Wbp, I_tmo_Wg, I_tmo_Wo,
  I_tmo_ln2b, I_tmo_ln2g, I_tmo_lnb, I_tmo_lng,
  N_PARAMS
};

static const long PSZ[N_PARAMS] = {
  65536, 65536, 65536, 128, 128,                                  // pt
  2048, 196608, 196608, 196608, 196608, 196608, 384, 384, 128, 128, // sa
  589824, 589824, 589824, 384, 384,                               // st
  512, 16384, 16384, 16384, 16384, 16384, 128, 128,               // tae
  512, 16384, 16384, 16384, 16384, 16384, 128, 128,               // tas
  16384, 16384, 16384, 16384, 16384, 16384, 128, 128, 128, 128,   // tmi
  16384, 16384, 16384, 16384, 16384, 16384, 128, 128, 128, 128    // tmo
};

static void gemmL(hipStream_t st, int epi, bool w16,
                  const __bf16* A, const float* Wf, const __bf16* Wh,
                  const float* W2, float* oF, __bf16* oH,
                  const float* mul, const float* res,
                  int M, int N, int K, int nb, long sA, long sW, long sC) {
  dim3 g((N + 63) / 64, (M + 63) / 64, nb), b(128);
  if (w16) {
    k_gemm<0, true><<<g, b, 0, st>>>(A, nullptr, Wh, nullptr, oF, oH, mul, res,
                                     M, N, K, sA, sW, sC);
    return;
  }
  switch (epi) {
    case 0: k_gemm<0, false><<<g, b, 0, st>>>(A, Wf, nullptr, W2, oF, oH, mul, res, M, N, K, sA, sW, sC); break;
    case 1: k_gemm<1, false><<<g, b, 0, st>>>(A, Wf, nullptr, W2, oF, oH, mul, res, M, N, K, sA, sW, sC); break;
    case 2: k_gemm<2, false><<<g, b, 0, st>>>(A, Wf, nullptr, W2, oF, oH, mul, res, M, N, K, sA, sW, sC); break;
    default: k_gemm<3, false><<<g, b, 0, st>>>(A, Wf, nullptr, W2, oF, oH, mul, res, M, N, K, sA, sW, sC); break;
  }
}

extern "C" void kernel_launch(void* const* d_in, const int* in_sizes, int n_in,
                              void* d_out, int out_size, void* d_ws, size_t ws_size,
                              hipStream_t stream) {
  (void)in_sizes; (void)n_in; (void)out_size; (void)ws_size;

  // ---- workspace layout ----
  char* w = (char*)d_ws;
  long off = 0;
  auto carve = [&](long bytes) -> char* {
    char* p = w + off;
    off += (bytes + 255) & ~255L;
    return p;
  };
  float*  Z    = (float*) carve(MPr * CPc * 4);   // pair state
  float*  S    = (float*) carve(MSr * CSc * 4);   // single state
  __bf16* H0   = (__bf16*)carve(MPr * CPc * 2);   // zl16
  __bf16* H1   = (__bf16*)carve(MPr * CPc * 2);   // a16 / q16
  __bf16* H2   = (__bf16*)carve(MPr * CPc * 2);   // b16 / k16
  __bf16* H3   = (__bf16*)carve(MPr * CPc * 2);   // v16
  __bf16* H4   = (__bf16*)carve(MPr * CPc * 2);   // ln2(x)16 / o16 / sl16
  float*  G    = (float*) carve(MPr * CPc * 4);   // gates / tae delta
  float*  X    = (float*) carve(MPr * CPc * 4);   // x f32 / o f32
  char*   TH   =          carve(52428800L);       // transition hidden / einsum bufs
  float*  BIA  = (float*) carve(4194304L);        // attention biases
  __bf16* ctA  = (__bf16*)TH;
  __bf16* ctB  = (__bf16*)(TH + 13107200L);
  float*  XBC  = (float*) (TH + 26214400L);
  __bf16* HID  = (__bf16*)TH;

  hipMemcpyAsync(S, d_in[0], MSr * CSc * 4, hipMemcpyDeviceToDevice, stream);
  hipMemcpyAsync(Z, d_in[1], MPr * CPc * 4, hipMemcpyDeviceToDevice, stream);
  const unsigned char* mask = (const unsigned char*)d_in[2];

  const int PG = 25600;   // grid for MP*CP elementwise (6.55M / 256)

  for (int l = 0; l < 2; ++l) {
    auto P = [&](int i) -> const float* {
      return (const float*)d_in[3 + i] + (long)l * PSZ[i];
    };
    auto ln = [&](const float* x, const float* g, const float* b, __bf16* o,
                  long rows, int C, int tr) {
      k_layernorm<<<(int)((rows + 3) / 4), 128, 0, stream>>>(
          x, g, b, o, (int)rows, C, Nc, tr);
    };

    // ================= TriangleMultiplication (outgoing, then incoming) ====
    for (int dir = 0; dir < 2; ++dir) {
      int ba = (dir == 0) ? I_tmo_Wag : I_tmi_Wag;
      // +0 Wag +1 Wap +2 Wbg +3 Wbp +4 Wg +5 Wo +6 ln2b +7 ln2g +8 lnb +9 lng
      ln(Z, P(ba + 9), P(ba + 8), H0, MPr, CPc, 0);
      gemmL(stream, 2, false, H0, P(ba + 0), nullptr, P(ba + 1),
            nullptr, H1, nullptr, nullptr, (int)MPr, CTc, CPc, 1, 0, 0, 0);
      gemmL(stream, 2, false, H0, P(ba + 2), nullptr, P(ba + 3),
            nullptr, H2, nullptr, nullptr, (int)MPr, CTc, CPc, 1, 0, 0, 0);
      gemmL(stream, 1, false, H0, P(ba + 4), nullptr, nullptr,
            G, nullptr, nullptr, nullptr, (int)MPr, CPc, CPc, 1, 0, 0, 0);
      int swapA = (dir == 0) ? 0 : 1, swapB = (dir == 0) ? 1 : 0;
      k_pack_tri<<<PG, 256, 0, stream>>>(H1, ctA, swapA);
      k_pack_tri<<<PG, 256, 0, stream>>>(H2, ctB, swapB);
      // batched [160x160x160] GEMM over b*CT = 256 channel batches
      gemmL(stream, 0, true, ctA, nullptr, ctB, nullptr,
            XBC, nullptr, nullptr, nullptr, Nc, Nc, Nc, Bc * CTc,
            25600, 25600, 25600);
      k_unpack_tri<<<PG, 256, 0, stream>>>(XBC, X);
      ln(X, P(ba + 7), P(ba + 6), H4, MPr, CTc, 0);
      gemmL(stream, 0, false, H4, P(ba + 5), nullptr, nullptr,
            Z, nullptr, G, Z, (int)MPr, CPc, CTc, 1, 0, 0, 0);
    }

    // ================= TriangleAttention (starting, then ending) ===========
    for (int dir = 0; dir < 2; ++dir) {
      int ba = (dir == 0) ? I_tas_Wb : I_tae_Wb;
      // +0 Wb +1 Wg +2 Wk +3 Wo +4 Wq +5 Wv +6 lnb +7 lng
      int tr = (dir == 0) ? 0 : 1;
      ln(Z, P(ba + 7), P(ba + 6), H0, MPr, CPc, tr);
      gemmL(stream, 0, false, H0, P(ba + 4), nullptr, nullptr,
            nullptr, H1, nullptr, nullptr, (int)MPr, HTc * DTc, CPc, 1, 0, 0, 0);
      gemmL(stream, 0, false, H0, P(ba + 2), nullptr, nullptr,
            nullptr, H2, nullptr, nullptr, (int)MPr, HTc * DTc, CPc, 1, 0, 0, 0);
      gemmL(stream, 0, false, H0, P(ba + 5), nullptr, nullptr,
            nullptr, H3, nullptr, nullptr, (int)MPr, HTc * DTc, CPc, 1, 0, 0, 0);
      gemmL(stream, 0, false, H0, P(ba + 0), nullptr, nullptr,
            BIA, nullptr, nullptr, nullptr, (int)MPr, HTc, CPc, 1, 0, 0, 0);
      gemmL(stream, 1, false, H0, P(ba + 1), nullptr, nullptr,
            G, nullptr, nullptr, nullptr, (int)MPr, HTc * DTc, CPc, 1, 0, 0, 0);
      k_tri_attn<<<dim3(10, Nc, Bc * HTc), 64, 0, stream>>>(H1, H2, H3, BIA, X);
      k_mul_to_bf16<<<PG, 256, 0, stream>>>(X, G, H4, MPr * CPc);
      if (dir == 0) {
        gemmL(stream, 0, false, H4, P(ba + 3), nullptr, nullptr,
              Z, nullptr, nullptr, Z, (int)MPr, CPc, HTc * DTc, 1, 0, 0, 0);
      } else {
        gemmL(stream, 0, false, H4, P(ba + 3), nullptr, nullptr,
              G, nullptr, nullptr, nullptr, (int)MPr, CPc, HTc * DTc, 1, 0, 0, 0);
        k_add_transpose<<<PG, 256, 0, stream>>>(Z, G);
      }
    }

    // ================= Pair transition (SwiGLU n=4) ========================
    ln(Z, P(I_pt_lng), P(I_pt_lnb), H0, MPr, CPc, 0);
    gemmL(stream, 3, false, H0, P(I_pt_W1), nullptr, P(I_pt_W2),
          nullptr, HID, nullptr, nullptr, (int)MPr, 4 * CPc, CPc, 1, 0, 0, 0);
    gemmL(stream, 0, false, HID, P(I_pt_W3), nullptr, nullptr,
          Z, nullptr, nullptr, Z, (int)MPr, CPc, 4 * CPc, 1, 0, 0, 0);

    // ================= AttentionPairBias (single rep) ======================
    ln(S, P(I_sa_lnsg), P(I_sa_lnsb), H4, MSr, CSc, 0);
    ln(Z, P(I_sa_lnzg), P(I_sa_lnzb), H0, MPr, CPc, 0);
    gemmL(stream, 0, false, H4, P(I_sa_Wq), nullptr, nullptr,
          nullptr, H1, nullptr, nullptr, (int)MSr, HSc * DSc, CSc, 1, 0, 0, 0);
    gemmL(stream, 0, false, H4, P(I_sa_Wk), nullptr, nullptr,
          nullptr, H2, nullptr, nullptr, (int)MSr, HSc * DSc, CSc, 1, 0, 0, 0);
    gemmL(stream, 0, false, H4, P(I_sa_Wv), nullptr, nullptr,
          nullptr, H3, nullptr, nullptr, (int)MSr, HSc * DSc, CSc, 1, 0, 0, 0);
    gemmL(stream, 0, false, H0, P(I_sa_Wb), nullptr, nullptr,
          BIA, nullptr, nullptr, nullptr, (int)MPr, HSc, CPc, 1, 0, 0, 0);
    gemmL(stream, 1, false, H4, P(I_sa_Wg), nullptr, nullptr,
          G, nullptr, nullptr, nullptr, (int)MSr, HSc * DSc, CSc, 1, 0, 0, 0);
    k_single_attn<<<dim3(10, 1, Bc * HSc), 64, 0, stream>>>(H1, H2, H3, BIA, mask, X);
    k_mul_to_bf16<<<(int)((MSr * HSc * DSc + 255) / 256), 256, 0, stream>>>(
        X, G, H4, MSr * HSc * DSc);
    gemmL(stream, 0, false, H4, P(I_sa_Wo), nullptr, nullptr,
          S, nullptr, nullptr, S, (int)MSr, CSc, HSc * DSc, 1, 0, 0, 0);

    // ================= Single transition ===================================
    ln(S, P(I_st_lng), P(I_st_lnb), H4, MSr, CSc, 0);
    gemmL(stream, 3, false, H4, P(I_st_W1), nullptr, P(I_st_W2),
          nullptr, HID, nullptr, nullptr, (int)MSr, 4 * CSc, CSc, 1, 0, 0, 0);
    gemmL(stream, 0, false, HID, P(I_st_W3), nullptr, nullptr,
          S, nullptr, nullptr, S, (int)MSr, CSc, 4 * CSc, 1, 0, 0, 0);
  }

  // outputs: (single, pair) concatenated flat
  hipMemcpyAsync(d_out, S, MSr * CSc * 4, hipMemcpyDeviceToDevice, stream);
  hipMemcpyAsync((float*)d_out + MSr * CSc, Z, MPr * CPc * 4,
                 hipMemcpyDeviceToDevice, stream);
}
